// RotaryCrossAttention_67207648247943
// MI455X (gfx1250) — matrix operand order repaired and checked by reading, not compiled, not run on hardware
//
#include <hip/hip_runtime.h>

typedef __attribute__((ext_vector_type(16))) __bf16 v16bf;
typedef __attribute__((ext_vector_type(8)))  __bf16 v8bf;
typedef __attribute__((ext_vector_type(4)))  __bf16 v4bf;
typedef __attribute__((ext_vector_type(8)))  float  v8f;

#define DMODEL 768
#define NHEADS 12
#define HDIM   64
#define BATCH  2
#define TQ     2048
#define TKV    4096

static __device__ __forceinline__ v8f wmma_bf16(v16bf a, v16bf b, v8f c) {
  // (neg_a, A, neg_b, B, c_mod, C, reuse_a, reuse_b)
  return __builtin_amdgcn_wmma_f32_16x16x32_bf16(false, a, false, b, (short)0, c, false, false);
}

// A-operand gather: two 16-byte chunks (callers use contiguous K: k = 16*half + s, as in B)
static __device__ __forceinline__ v16bf ld_a16(const __bf16* base, int off0, int off1) {
  union { v16bf v; v8bf h[2]; } u;
  u.h[0] = *(const v8bf*)(base + off0);
  u.h[1] = *(const v8bf*)(base + off1);
  return u.v;
}
// B-operand: 16 contiguous K values per half-wave
static __device__ __forceinline__ v16bf ld_b16(const __bf16* p) {
  return *(const v16bf*)p;
}

static __device__ __forceinline__ float hmax16(float v) {
  v = fmaxf(v, __shfl_xor(v, 1, 32));
  v = fmaxf(v, __shfl_xor(v, 2, 32));
  v = fmaxf(v, __shfl_xor(v, 4, 32));
  v = fmaxf(v, __shfl_xor(v, 8, 32));
  return v;
}
static __device__ __forceinline__ float hsum16(float v) {
  v += __shfl_xor(v, 1, 32);
  v += __shfl_xor(v, 2, 32);
  v += __shfl_xor(v, 4, 32);
  v += __shfl_xor(v, 8, 32);
  return v;
}

// async copy of one 16-byte segment global -> LDS (tracked by ASYNCcnt)
static __device__ __forceinline__ void async_b128(const __bf16* src, void* lds_dst) {
  const unsigned dst = (unsigned)(uintptr_t)lds_dst;  // low 32 bits = LDS address
  asm volatile("global_load_async_to_lds_b128 %0, %1, off"
               :: "v"(dst), "v"(src) : "memory");
}

// ---------------------------------------------------------------------------
// One-time fp32 -> bf16 conversion (vectorized). n must be a multiple of 4.
// ---------------------------------------------------------------------------
__global__ void cvt_f32_to_bf16(const float* __restrict__ X, __bf16* __restrict__ Y,
                                int n) {
  const int idx = (blockIdx.x * blockDim.x + threadIdx.x) * 4;
  if (idx >= n) return;
  const float4 f = *(const float4*)(X + idx);
  v4bf o;
  o[0] = (__bf16)f.x; o[1] = (__bf16)f.y; o[2] = (__bf16)f.z; o[3] = (__bf16)f.w;
  *(v4bf*)(Y + idx) = o;
}

// ---------------------------------------------------------------------------
// C[M x 768] = A[M x 768] @ W^T + bias.  A, W bf16; out bf16 or fp32.
// One wave = 16x64 output tile (4 f32 accumulators), blockDim=128, grid=(M/16,3).
// ---------------------------------------------------------------------------
template <typename TOUT>
__global__ __launch_bounds__(128) void gemm_bias_wmma(
    const __bf16* __restrict__ A, const __bf16* __restrict__ W,
    const float* __restrict__ bias, TOUT* __restrict__ C) {
  const int lane = threadIdx.x & 31;
  const int wv   = threadIdx.x >> 5;
  const int half = lane >> 4;
  const int l16  = lane & 15;
  const int mbase = blockIdx.x * 16;
  const int nbase = blockIdx.y * 256 + wv * 64;

  v8f acc[4];
#pragma unroll
  for (int t = 0; t < 4; ++t) {
    const float bv = bias[nbase + t * 16 + l16];
#pragma unroll
    for (int r = 0; r < 8; ++r) acc[t][r] = bv;
  }

  const __bf16* arow = A + (size_t)(mbase + l16) * DMODEL;
  const int aoff0 = half * 16;
  const int aoff1 = half * 16 + 8;
  const __bf16* wrow[4];
#pragma unroll
  for (int t = 0; t < 4; ++t)
    wrow[t] = W + (size_t)(nbase + t * 16 + l16) * DMODEL + half * 16;

  for (int k0 = 0; k0 < DMODEL; k0 += 32) {
    const v16bf a = ld_a16(arow + k0, aoff0, aoff1);
#pragma unroll
    for (int t = 0; t < 4; ++t)
      acc[t] = wmma_bf16(a, ld_b16(wrow[t] + k0), acc[t]);
  }

#pragma unroll
  for (int t = 0; t < 4; ++t)
#pragma unroll
    for (int r = 0; r < 8; ++r)
      C[(size_t)(mbase + r + 8 * half) * DMODEL + nbase + t * 16 + l16] =
          (TOUT)acc[t][r];
}

// ---------------------------------------------------------------------------
// Rotary per reference slicing of emb = concat(sin, cos):
//   sin_i = emb[2i], cos_i = emb[2i+1];
//   out[i] = x[2i]*cos_i - x[2i+1]*sin_i; out[i+32] = x[2i]*sin_i + x[2i+1]*cos_i
// X: [B, T, 768] bf16 -> Y: [B, H, T, 64] bf16, scaled by `scale`
// (scale = 1/sqrt(64) = 2^-3 for Q: exact exponent shift in bf16).
// ---------------------------------------------------------------------------
__global__ void rotary_bf16(const __bf16* __restrict__ X, __bf16* __restrict__ Y,
                            int T, float scale) {
  const int idx = blockIdx.x * blockDim.x + threadIdx.x;
  if (idx >= BATCH * T * NHEADS * 32) return;
  const int i = idx & 31;
  int rem = idx >> 5;
  const int h = rem % NHEADS; rem /= NHEADS;
  const int t = rem % T;
  const int b = rem / T;

  const __bf16* x = X + (size_t)(b * T + t) * DMODEL + h * HDIM;
  const float x1 = (float)x[2 * i], x2 = (float)x[2 * i + 1];
  const float tt = (float)t;
  const float LF = 0.28782313662425575f;  // ln(10000)/32
  int j = 2 * i;
  const float s = (j < 32) ? __sinf(tt * __expf(-LF * j))
                           : __cosf(tt * __expf(-LF * (j - 32)));
  j = 2 * i + 1;
  const float c = (j < 32) ? __sinf(tt * __expf(-LF * j))
                           : __cosf(tt * __expf(-LF * (j - 32)));
  __bf16* y = Y + (size_t)((b * NHEADS + h) * T + t) * HDIM;
  y[i]      = (__bf16)((x1 * c - x2 * s) * scale);
  y[i + 32] = (__bf16)((x1 * s + x2 * c) * scale);
}

// X: [B, Tk, 768] bf16 -> Y: [B, H, 64, Tk] bf16 (V transposed for PV B-operand)
__global__ void v_transpose_bf16(const __bf16* __restrict__ X, __bf16* __restrict__ Y) {
  const int idx = blockIdx.x * blockDim.x + threadIdx.x;
  if (idx >= BATCH * TKV * DMODEL) return;
  const int f = idx % DMODEL;
  int rem = idx / DMODEL;
  const int t = rem % TKV;
  const int b = rem / TKV;
  const int h = f / HDIM, d = f % HDIM;
  Y[((size_t)(b * NHEADS + h) * HDIM + d) * TKV + t] = X[idx];
}

// ---------------------------------------------------------------------------
// Flash attention. Block = 4 waves on the SAME (b,h), 4 consecutive 16-row
// q tiles. K/V streamed in 64-key chunks staged ONCE per block into LDS via
// async global->LDS DMA (double-buffered: async engine fetches chunk i+1
// while WMMAs consume chunk i). Q pre-scaled by 1/8.
// Q: [B,H,Tq,64], K: [B,H,Tk,64], Vt: [B,H,64,Tk] bf16. O: [B,Tq,768] bf16.
// ---------------------------------------------------------------------------
__global__ __launch_bounds__(128) void flash_attn_wmma(
    const __bf16* __restrict__ Q, const __bf16* __restrict__ K,
    const __bf16* __restrict__ Vt, __bf16* __restrict__ O) {
  __shared__ __align__(16) __bf16 kbuf[2][64][64];  // [buf][key][d]   8KB each
  __shared__ __align__(16) __bf16 vbuf[2][64][64];  // [buf][d][key]   8KB each
  __shared__ __align__(16) __bf16 pls[4][16][64];   // per-wave P tile 8KB
  const int tid  = threadIdx.x;
  const int lane = tid & 31;
  const int wv   = tid >> 5;
  const int half = lane >> 4;
  const int l16  = lane & 15;

  const int gq  = blockIdx.x;               // B*H*(TQ/64) blocks
  const int qt4 = gq % (TQ / 64);
  const int h   = (gq / (TQ / 64)) % NHEADS;
  const int b   = gq / ((TQ / 64) * NHEADS);
  const int qtile = qt4 * 4 + wv;

  const __bf16* qp = Q + ((size_t)(b * NHEADS + h) * TQ + qtile * 16) * HDIM;
  const __bf16* kp = K + (size_t)(b * NHEADS + h) * TKV * HDIM;
  const __bf16* vp = Vt + (size_t)(b * NHEADS + h) * HDIM * TKV;

  const int aoff0 = half * 16;
  const int aoff1 = half * 16 + 8;

  // cooperative async stage of one 64-key K and V chunk (8 async ops per wave)
  auto stage = [&](int buf, int kb) {
#pragma unroll
    for (int it = 0; it < 4; ++it) {
      const int s = tid + it * 128;           // 512 16B segments
      const int row = s >> 3, seg = (s & 7) * 8;
      async_b128(kp + (size_t)(kb + row) * HDIM + seg, &kbuf[buf][row][seg]);
    }
#pragma unroll
    for (int it = 0; it < 4; ++it) {
      const int s = tid + it * 128;
      const int row = s >> 3, seg = (s & 7) * 8;
      async_b128(vp + (size_t)row * TKV + kb + seg, &vbuf[buf][row][seg]);
    }
  };

  // Q A-operands: contraction over D=64 -> two 16x32 A tiles
  const __bf16* qrow = qp + (size_t)l16 * HDIM;
  const v16bf qa0 = ld_a16(qrow, aoff0, aoff1);
  const v16bf qa1 = ld_a16(qrow + 32, aoff0, aoff1);

  v8f acc[4] = {};
  float mrow[8], lrow[8];
#pragma unroll
  for (int r = 0; r < 8; ++r) { mrow[r] = -1e30f; lrow[r] = 0.f; }

  stage(0, 0);
  for (int kb = 0; kb < TKV; kb += 64) {
    const int buf = (kb >> 6) & 1;
    if (kb + 64 < TKV) {
      stage(buf ^ 1, kb + 64);  // prefetch next chunk via async DMA
      // async loads retire in order: <=8 outstanding => previous stage done
      asm volatile("s_wait_asynccnt 8" ::: "memory");
    } else {
      asm volatile("s_wait_asynccnt 0" ::: "memory");
    }
    __syncthreads();  // all waves' staged data visible

    // S = q.k^T for 64 keys from LDS: 4 C-tiles, each 2 WMMAs (K = D = 64)
    v8f s[4];
#pragma unroll
    for (int n = 0; n < 4; ++n) {
      const __bf16* kr = &kbuf[buf][n * 16 + l16][half * 16];
      v8f t = {};
      t = wmma_bf16(qa0, ld_b16(kr), t);
      t = wmma_bf16(qa1, ld_b16(kr + 32), t);
      s[n] = t;
    }

    // online softmax; row r + 8*half lives in element r, striped over a half-wave
#pragma unroll
    for (int r = 0; r < 8; ++r) {
      const float a0 = s[0][r], a1 = s[1][r], a2 = s[2][r], a3 = s[3][r];
      const float mx   = hmax16(fmaxf(fmaxf(a0, a1), fmaxf(a2, a3)));
      const float mnew = fmaxf(mrow[r], mx);
      const float corr = __expf(mrow[r] - mnew);
      const float p0 = __expf(a0 - mnew);
      const float p1 = __expf(a1 - mnew);
      const float p2 = __expf(a2 - mnew);
      const float p3 = __expf(a3 - mnew);
      const float rs = hsum16((p0 + p1) + (p2 + p3));
      lrow[r] = lrow[r] * corr + rs;
      mrow[r] = mnew;
#pragma unroll
      for (int t = 0; t < 4; ++t) acc[t][r] *= corr;
      __bf16* pr = &pls[wv][r + 8 * half][0];
      pr[l16]      = (__bf16)p0;
      pr[16 + l16] = (__bf16)p1;
      pr[32 + l16] = (__bf16)p2;
      pr[48 + l16] = (__bf16)p3;
    }
    asm volatile("s_wait_dscnt 0" ::: "memory");  // DS in-order per wave
    const __bf16* prow = &pls[wv][l16][0];
    const v16bf pa0 = ld_a16(prow, aoff0, aoff1);
    const v16bf pa1 = ld_a16(prow + 32, aoff0, aoff1);
    asm volatile("" ::: "memory");

    // O += P @ V from LDS: 4 d-tiles x 2 key-subchunks
#pragma unroll
    for (int t = 0; t < 4; ++t) {
      const __bf16* vr = &vbuf[buf][t * 16 + l16][half * 16];
      acc[t] = wmma_bf16(pa0, ld_b16(vr), acc[t]);
      acc[t] = wmma_bf16(pa1, ld_b16(vr + 32), acc[t]);
    }
    __syncthreads();  // everyone done reading buf before it is re-staged
  }

  // normalize rows, scatter to [B, Tq, H*64]
  __bf16* op = O + (size_t)(b * TQ + qtile * 16) * DMODEL + h * HDIM;
#pragma unroll
  for (int r = 0; r < 8; ++r) {
    const float inv = 1.0f / lrow[r];
#pragma unroll
    for (int t = 0; t < 4; ++t)
      op[(size_t)(r + 8 * half) * DMODEL + t * 16 + l16] = (__bf16)(acc[t][r] * inv);
  }
}

// ---------------------------------------------------------------------------
extern "C" void kernel_launch(void* const* d_in, const int* in_sizes, int n_in,
                              void* d_out, int out_size, void* d_ws, size_t ws_size,
                              hipStream_t stream) {
  (void)in_sizes; (void)n_in; (void)out_size; (void)ws_size;
  const float* query = (const float*)d_in[0];
  const float* key   = (const float*)d_in[1];
  const float* value = (const float*)d_in[2];
  const float* Wq = (const float*)d_in[3];
  const float* bq = (const float*)d_in[4];
  const float* Wk = (const float*)d_in[5];
  const float* bk = (const float*)d_in[6];
  const float* Wv = (const float*)d_in[7];
  const float* bv = (const float*)d_in[8];
  const float* Wo = (const float*)d_in[9];
  const float* bo = (const float*)d_in[10];

  char* ws = (char*)d_ws;
  size_t off = 0;
  auto alloc = [&](size_t bytes) -> void* {
    void* p = ws + off;
    off += (bytes + 255) & ~(size_t)255;
    return p;
  };
  const size_t NQ = (size_t)BATCH * TQ * DMODEL;
  const size_t NK = (size_t)BATCH * TKV * DMODEL;
  const size_t NW = (size_t)DMODEL * DMODEL;

  __bf16* qx = (__bf16*)alloc(NQ * 2);   // bf16 inputs
  __bf16* kx = (__bf16*)alloc(NK * 2);
  __bf16* vx = (__bf16*)alloc(NK * 2);
  __bf16* wq = (__bf16*)alloc(NW * 2);   // bf16 weights
  __bf16* wk = (__bf16*)alloc(NW * 2);
  __bf16* wvw = (__bf16*)alloc(NW * 2);
  __bf16* wo = (__bf16*)alloc(NW * 2);
  __bf16* pq = (__bf16*)alloc(NQ * 2);   // projected q/k/v (bf16)
  __bf16* pk = (__bf16*)alloc(NK * 2);
  __bf16* pv = (__bf16*)alloc(NK * 2);
  __bf16* qb = (__bf16*)alloc(NQ * 2);   // rotary q [B,H,Tq,64] (pre-scaled 1/8)
  __bf16* kb = (__bf16*)alloc(NK * 2);   // rotary k [B,H,Tk,64]
  __bf16* vt = (__bf16*)alloc(NK * 2);   // v^T [B,H,64,Tk]
  __bf16* ab = (__bf16*)alloc(NQ * 2);   // attention out [B,Tq,768]

  const dim3 blk(128);
  auto cvt = [&](const float* src, __bf16* dst, size_t n) {
    cvt_f32_to_bf16<<<(unsigned)((n / 4 + 255) / 256), 256, 0, stream>>>(src, dst, (int)n);
  };
  cvt(query, qx, NQ); cvt(key, kx, NK); cvt(value, vx, NK);
  cvt(Wq, wq, NW); cvt(Wk, wk, NW); cvt(Wv, wvw, NW); cvt(Wo, wo, NW);

  // input projections (bf16 WMMA, f32 accumulate, bf16 out)
  gemm_bias_wmma<__bf16><<<dim3(BATCH * TQ / 16, 3),  blk, 0, stream>>>(qx, wq, bq, pq);
  gemm_bias_wmma<__bf16><<<dim3(BATCH * TKV / 16, 3), blk, 0, stream>>>(kx, wk, bk, pk);
  gemm_bias_wmma<__bf16><<<dim3(BATCH * TKV / 16, 3), blk, 0, stream>>>(vx, wvw, bv, pv);

  // rotary + head-major layout; fold 1/sqrt(64) into q (exact in bf16)
  {
    const int nq = BATCH * TQ * NHEADS * 32;
    rotary_bf16<<<(nq + 255) / 256, 256, 0, stream>>>(pq, qb, TQ, 0.125f);
    const int nk = BATCH * TKV * NHEADS * 32;
    rotary_bf16<<<(nk + 255) / 256, 256, 0, stream>>>(pk, kb, TKV, 1.0f);
    const int nv = BATCH * TKV * DMODEL;
    v_transpose_bf16<<<(nv + 255) / 256, 256, 0, stream>>>(pv, vt);
  }

  // flash attention: 4 waves/block share (b,h) and the async-staged K/V chunks
  {
    const int nblocks = BATCH * NHEADS * (TQ / 64);  // 768
    flash_attn_wmma<<<nblocks, blk, 0, stream>>>(qb, kb, vt, ab);
  }

  // output projection (f32 out to d_out)
  gemm_bias_wmma<float><<<dim3(BATCH * TQ / 16, 3), blk, 0, stream>>>(ab, wo, bo, (float*)d_out);
}
